// DxPrivacy_19593640805104
// MI455X (gfx1250) — matrix operand order repaired and checked by reading, not compiled, not run on hardware
//
#include <hip/hip_runtime.h>

// Problem sizes from the reference
#define DIM 1024       // D
#define VOCAB 32000    // V
#define MROWS 4096     // B*S

typedef __attribute__((ext_vector_type(16))) __bf16 v16bf;
typedef __attribute__((ext_vector_type(8)))  float  v8f;

__device__ __forceinline__ unsigned short f2bf(float f) {
    unsigned u = __float_as_uint(f);
    u += 0x7FFFu + ((u >> 16) & 1u);   // round-to-nearest-even
    return (unsigned short)(u >> 16);
}

// Kernel 1: fp32 table -> bf16 table in workspace (L2-resident afterwards)
__global__ void __launch_bounds__(256)
cvt_table_kernel(const float* __restrict__ table, unsigned short* __restrict__ tbf) {
    size_t i = ((size_t)blockIdx.x * 256 + threadIdx.x) * 4;   // grid covers V*D exactly
    float4 f = *(const float4*)(table + i);
    unsigned p0 = (unsigned)f2bf(f.x) | ((unsigned)f2bf(f.y) << 16);
    unsigned p1 = (unsigned)f2bf(f.z) | ((unsigned)f2bf(f.w) << 16);
    *(uint2*)(tbf + i) = make_uint2(p0, p1);
}

// Kernel 2: fused (embeds+noise) -> bf16 A panel in LDS -> WMMA similarity GEMM
//           -> running row argmax -> gather winning fp32 table row.
// One workgroup (8 waves) per 16 query rows. Each wave: A cached one octet
// (8 K-steps, 64 VGPRs) at a time; B software-pipelined depth-2 (bcur/bnxt,
// 64 VGPRs) so each 4-WMMA group executes with 8 global_load_b128 in flight.
__global__ void __launch_bounds__(256)
nearest_kernel(const float* __restrict__ embeds,
               const float* __restrict__ noise,
               const float* __restrict__ table,
               const unsigned short* __restrict__ tbf,
               float* __restrict__ out) {
    __shared__ unsigned short ldsA[16 * DIM];   // 32 KB bf16 A panel
    __shared__ float redv[8][16];
    __shared__ int   redi[8][16];
    __shared__ int   rowIdx[16];

    const int tid = threadIdx.x;
    const int wgM = blockIdx.x * 16;            // first query row of this WG

    // ---- Stage A = bf16(embeds + noise) rows [wgM, wgM+16) into LDS ----
    {
        const float4* e4 = (const float4*)(embeds + (size_t)wgM * DIM);
        const float4* n4 = (const float4*)(noise  + (size_t)wgM * DIM);
        for (int i = tid; i < (16 * DIM) / 4; i += 256) {
            float4 e = e4[i];
            float4 n = n4[i];
            unsigned p0 = (unsigned)f2bf(e.x + n.x) | ((unsigned)f2bf(e.y + n.y) << 16);
            unsigned p1 = (unsigned)f2bf(e.z + n.z) | ((unsigned)f2bf(e.w + n.w) << 16);
            *(uint2*)(ldsA + 4 * i) = make_uint2(p0, p1);
        }
    }
    __syncthreads();

    const int lane  = tid & 31;
    const int wave  = tid >> 5;
    const int l15   = lane & 15;     // A row (M) for this lane; also C/D column N
    const int nhalf = lane >> 4;     // which K-half this lane holds

    // A operand in the contiguous K order used here: lane m (0-15) holds K = k0+{0..15},
    // lane m+16 holds K = k0+{16..31}.
    const unsigned short* aBase = ldsA + l15 * DIM + nhalf * 16;

    float bestv[8];
    int   besti[8];
#pragma unroll
    for (int r = 0; r < 8; ++r) { bestv[r] = -3.402823466e38f; besti[r] = 0; }

    const __bf16* Tb = (const __bf16*)tbf;

    // 2000 column tiles of 16 -> 500 groups of 4 tiles, 8 waves round-robin.
#pragma unroll 1
    for (int tg = wave; tg < (VOCAB / 64); tg += 8) {
        const int n0 = tg * 64 + l15;          // vocab row for acc[0] (lane's N)
        v8f acc[4];
        const __bf16* bp[4];
#pragma unroll
        for (int j = 0; j < 4; ++j) {
            acc[j] = (v8f){};
            // B operand, same contiguous K order as A: lane n holds 16 contiguous K values of vocab row
            // n, K offset 0 (lanes 0-15) or 16 (lanes 16-31).
            bp[j] = Tb + (size_t)(n0 + 16 * j) * DIM + nhalf * 16;
        }

        // Prime the B pipeline with K-step 0.
        v16bf bcur[4];
#pragma unroll
        for (int j = 0; j < 4; ++j) bcur[j] = *(const v16bf*)(bp[j]);

        // K = 1024 in 4 octets of 8 K-steps; A octet cached in 64 VGPRs.
#pragma unroll 1
        for (int k8 = 0; k8 < 4; ++k8) {
            const int kbase = k8 * 8;
            v16bf af[8];
#pragma unroll
            for (int kk = 0; kk < 8; ++kk) {
                union { v16bf v; uint4 q[2]; } t;
                t.q[0] = *(const uint4*)(aBase + (kbase + kk) * 32);
                t.q[1] = *(const uint4*)(aBase + (kbase + kk) * 32 + 8);
                af[kk] = t.v;
            }
            // Steps 0..6: prefetch step kk+1 before consuming step kk.
#pragma unroll
            for (int kk = 0; kk < 7; ++kk) {
                v16bf bnxt[4];
#pragma unroll
                for (int j = 0; j < 4; ++j)
                    bnxt[j] = *(const v16bf*)(bp[j] + (kbase + kk + 1) * 32);
#pragma unroll
                for (int j = 0; j < 4; ++j)
                    acc[j] = __builtin_amdgcn_wmma_f32_16x16x32_bf16(
                                 false, af[kk], false, bcur[j], (short)0, acc[j],
                                 false, false);
#pragma unroll
                for (int j = 0; j < 4; ++j) bcur[j] = bnxt[j];
            }
            // Step 7: preload next octet's step 0 (if any), then consume.
            if (k8 < 3) {
                v16bf bnxt[4];
#pragma unroll
                for (int j = 0; j < 4; ++j)
                    bnxt[j] = *(const v16bf*)(bp[j] + (kbase + 8) * 32);
#pragma unroll
                for (int j = 0; j < 4; ++j)
                    acc[j] = __builtin_amdgcn_wmma_f32_16x16x32_bf16(
                                 false, af[7], false, bcur[j], (short)0, acc[j],
                                 false, false);
#pragma unroll
                for (int j = 0; j < 4; ++j) bcur[j] = bnxt[j];
            } else {
#pragma unroll
                for (int j = 0; j < 4; ++j)
                    acc[j] = __builtin_amdgcn_wmma_f32_16x16x32_bf16(
                                 false, af[7], false, bcur[j], (short)0, acc[j],
                                 false, false);
            }
        }

        // Running per-row argmax in C-layout registers.
        // VGPR r: rows M=r (lanes 0-15) / M=r+8 (lanes 16-31), N = l15.
#pragma unroll
        for (int j = 0; j < 4; ++j) {
            const int nj = n0 + 16 * j;
#pragma unroll
            for (int r = 0; r < 8; ++r) {
                if (acc[j][r] > bestv[r]) { bestv[r] = acc[j][r]; besti[r] = nj; }
            }
        }
    }

    // ---- Reduce argmax across the 16 lanes of each half-wave ----
#pragma unroll
    for (int off = 8; off; off >>= 1) {
#pragma unroll
        for (int r = 0; r < 8; ++r) {
            float ov = __shfl_xor(bestv[r], off, 16);
            int   oi = __shfl_xor(besti[r], off, 16);
            if (ov > bestv[r] || (ov == bestv[r] && oi < besti[r])) {
                bestv[r] = ov; besti[r] = oi;
            }
        }
    }
    if (l15 == 0) {
#pragma unroll
        for (int r = 0; r < 8; ++r) {
            redv[wave][nhalf * 8 + r] = bestv[r];
            redi[wave][nhalf * 8 + r] = besti[r];
        }
    }
    __syncthreads();

    // ---- Reduce across the 8 waves ----
    if (tid < 16) {
        float bv = redv[0][tid];
        int   bi = redi[0][tid];
#pragma unroll
        for (int w = 1; w < 8; ++w) {
            float v = redv[w][tid];
            int   i2 = redi[w][tid];
            if (v > bv || (v == bv && i2 < bi)) { bv = v; bi = i2; }
        }
        rowIdx[tid] = bi;
    }
    __syncthreads();

    // ---- Gather: copy winning fp32 table rows to output (4 KB/row) ----
#pragma unroll 1
    for (int r = 0; r < 16; ++r) {
        const float4* s = (const float4*)(table + (size_t)rowIdx[r] * DIM);
        float4*       d = (float4*)(out + (size_t)(wgM + r) * DIM);
        d[tid] = s[tid];   // 256 threads x 16 B = 1024 floats
    }
}

extern "C" void kernel_launch(void* const* d_in, const int* in_sizes, int n_in,
                              void* d_out, int out_size, void* d_ws, size_t ws_size,
                              hipStream_t stream) {
    const float* embeds = (const float*)d_in[0];   // [2,2048,1024] fp32
    const float* table  = (const float*)d_in[1];   // [32000,1024] fp32
    const float* noise  = (const float*)d_in[2];   // [2,2048,1024] fp32
    float* out = (float*)d_out;                    // [2,2048,1024] fp32

    unsigned short* tbf = (unsigned short*)d_ws;   // bf16 table, 65,536,000 B

    // Convert vocab table to bf16 (V*D/4 elements per thread-quad)
    cvt_table_kernel<<<(VOCAB * DIM) / (256 * 4), 256, 0, stream>>>(table, tbf);

    // Fused GEMM + argmax + gather: one WG per 16 query rows
    nearest_kernel<<<MROWS / 16, 256, 0, stream>>>(embeds, noise, table, tbf, out);
}
